// SelfAttentionLayer_29832842838526
// MI455X (gfx1250) — matrix operand order repaired and checked by reading, not compiled, not run on hardware
//
#include <hip/hip_runtime.h>

#define SEQ 4096
#define HID 1024
#define NEG_INF (-10000000000.0f)
#define PADROW 40   // bf16 elems per LDS tile row (80B) -> conflict-free b128 reads

typedef __bf16 bf16;
typedef __attribute__((ext_vector_type(16))) __bf16 bf16x16;
typedef __attribute__((ext_vector_type(8)))  __bf16 bf16x8;
typedef __attribute__((ext_vector_type(4)))  __bf16 bf16x4;
typedef __attribute__((ext_vector_type(8)))  float  f32x8;
typedef __attribute__((ext_vector_type(4)))  float  f32x4;
typedef __attribute__((ext_vector_type(4)))  int    v4i;

#if __has_builtin(__builtin_amdgcn_global_load_async_to_lds_b128) && \
    __has_builtin(__builtin_amdgcn_s_wait_asynccnt)
#define HAVE_ASYNC_LDS 1
#else
#define HAVE_ASYNC_LDS 0
#endif

// ---------------------------------------------------------------------------
// 16-byte global -> LDS copy (async on CDNA5; sync reg-copy fallback)
// Builtin signature (from hipcc diagnostic): (v4i as1*, v4i as3*, imm, imm)
// ---------------------------------------------------------------------------
__device__ __forceinline__ void copy16(const bf16* __restrict__ g, bf16* l) {
#if HAVE_ASYNC_LDS
  __builtin_amdgcn_global_load_async_to_lds_b128(
      (__attribute__((address_space(1))) v4i*)(bf16*)g,
      (__attribute__((address_space(3))) v4i*)l, 0, 0);
#else
  *(bf16x8*)l = *(const bf16x8*)g;
#endif
}

__device__ __forceinline__ void wait_async() {
#if HAVE_ASYNC_LDS
  __builtin_amdgcn_s_wait_asynccnt(0);
#endif
}

// ---------------------------------------------------------------------------
// WMMA fragment helpers (layouts per CDNA5 ISA §7.12.2, wave32)
// A (16x32 bf16): lane L -> row M=L&15; elems 0..15 = K(kb+16h..) contiguous (this code's k order)
// B (32x16 bf16): lane L -> col N=L&15; elems 0..15 = K(kb+16h..) contiguous (this code's order) when the
//                 operand is stored K-contiguous per column.
// C/D (16x16 f32): lane L -> N=L&15; VGPR r -> M = r + 8*(L>>4)
// ---------------------------------------------------------------------------
__device__ __forceinline__ bf16x16 frag16(const bf16* p0, const bf16* p1) {
  bf16x8 lo = *(const bf16x8*)p0;
  bf16x8 hi = *(const bf16x8*)p1;
  return __builtin_shufflevector(lo, hi, 0,1,2,3,4,5,6,7,8,9,10,11,12,13,14,15);
}

__device__ __forceinline__ f32x8 wmma_bf16(bf16x16 a, bf16x16 b, f32x8 c) {
  return __builtin_amdgcn_wmma_f32_16x16x32_bf16(false, a, false, b, (short)0, c,
                                                 false, false);
}

__device__ __forceinline__ bf16x16 load_a_f32(const float* __restrict__ base, int ldk,
                                              int row, int kb, int h) {
  const float* p = base + (size_t)row * ldk + kb + 16 * h;
  f32x4 c0 = *(const f32x4*)(p);
  f32x4 c1 = *(const f32x4*)(p + 4);
  f32x4 c2 = *(const f32x4*)(p + 8);
  f32x4 c3 = *(const f32x4*)(p + 12);
  bf16x16 r;
#pragma unroll
  for (int e = 0; e < 4; ++e) {
    r[e]      = (bf16)c0[e];
    r[e + 4]  = (bf16)c1[e];
    r[e + 8]  = (bf16)c2[e];
    r[e + 12] = (bf16)c3[e];
  }
  return r;
}

// ---------------------------------------------------------------------------
// Copy one 128x32 bf16 tile (global, row-major, ld=ldk) into padded LDS tile.
// 256 threads x 2 chunks x 16B = 8KB.
// ---------------------------------------------------------------------------
__device__ __forceinline__ void tile_copy(const bf16* __restrict__ G, int ldk,
                                          int rowBase, int kb, bf16* lds, int tid) {
#pragma unroll
  for (int i = 0; i < 2; ++i) {
    int c   = tid + i * 256;   // 0..511
    int row = c >> 2;          // 0..127
    int off = (c & 3) * 8;     // elem offset: 0,8,16,24
    copy16(G + (size_t)(rowBase + row) * ldk + kb + off, lds + row * PADROW + off);
  }
}

// ---------------------------------------------------------------------------
// Double-buffered LDS-staged bf16 GEMM inner loop. Both operands ld = HID,
// kSteps k-steps of 32. A rows from blockM, B "columns" (rows of the stored
// array) from blockN. Wave tile 64x32, WG tile 128x128 (8 waves as 2x4).
// ---------------------------------------------------------------------------
__device__ __forceinline__ void gemm_lds(const bf16* __restrict__ A,
                                         const bf16* __restrict__ B,
                                         int blockM, int blockN, int kSteps,
                                         bf16* As0, bf16* As1, bf16* Bs0, bf16* Bs1,
                                         f32x8 acc[4][2]) {
  const int tid  = threadIdx.x;
  const int lane = tid & 31;
  const int wave = tid >> 5;
  const int h  = lane >> 4;
  const int ln = lane & 15;
  const int wm = (wave >> 2) * 64;
  const int wn = (wave & 3) * 32;

  bf16* Abuf[2] = {As0, As1};
  bf16* Bbuf[2] = {Bs0, Bs1};

  tile_copy(A, HID, blockM, 0, Abuf[0], tid);
  tile_copy(B, HID, blockN, 0, Bbuf[0], tid);
  wait_async();
  __syncthreads();

  for (int ks = 0; ks < kSteps; ++ks) {
    const int cur = ks & 1;
    const int nxt = cur ^ 1;
    if (ks + 1 < kSteps) {  // prefetch next tiles into the other buffer
      tile_copy(A, HID, blockM, (ks + 1) * 32, Abuf[nxt], tid);
      tile_copy(B, HID, blockN, (ks + 1) * 32, Bbuf[nxt], tid);
    }

    const bf16* a = Abuf[cur];
    const bf16* b = Bbuf[cur];
    const bf16* pb0 = b + (wn + ln) * PADROW + 16 * h;
    const bf16* pb1 = b + (wn + 16 + ln) * PADROW + 16 * h;
    bf16x16 bb0 = frag16(pb0, pb0 + 8);
    bf16x16 bb1 = frag16(pb1, pb1 + 8);
#pragma unroll
    for (int i = 0; i < 4; ++i) {
      const bf16* pa = a + (wm + i * 16 + ln) * PADROW + 16 * h;
      bf16x16 aa = frag16(pa, pa + 8);
      acc[i][0] = wmma_bf16(aa, bb0, acc[i][0]);
      acc[i][1] = wmma_bf16(aa, bb1, acc[i][1]);
    }

    wait_async();       // next-tile async copies landed in LDS
    __syncthreads();    // also drains DScnt: cur-buffer reads done before reuse
  }
}

// ---------------------------------------------------------------------------
// fp32 -> bf16 conversion
// ---------------------------------------------------------------------------
__global__ __launch_bounds__(256) void cvt_kernel(const float* __restrict__ src,
                                                  bf16* __restrict__ dst, int n) {
  int i = (blockIdx.x * 256 + threadIdx.x) * 4;
  if (i < n) {
    f32x4 v = *(const f32x4*)(src + i);
    bf16x4 o;
#pragma unroll
    for (int e = 0; e < 4; ++e) o[e] = (bf16)v[e];
    *(bf16x4*)(dst + i) = o;
  }
}

// ---------------------------------------------------------------------------
// Projection: Y[SEQ,HID] = X[SEQ,HID] @ W[HID,HID]^T (bf16 in, f32 accum).
// transposeOut: write Yt[N][M] instead (V path -> clean B loads in av_kernel).
// ---------------------------------------------------------------------------
__global__ __launch_bounds__(256) void proj_kernel(const bf16* __restrict__ X,
                                                   const bf16* __restrict__ W,
                                                   bf16* __restrict__ Y,
                                                   bf16* __restrict__ Yt,
                                                   int transposeOut) {
  __shared__ bf16 As0[128 * PADROW], As1[128 * PADROW];
  __shared__ bf16 Bs0[128 * PADROW], Bs1[128 * PADROW];

  const int lane = threadIdx.x & 31;
  const int wave = threadIdx.x >> 5;
  const int h    = lane >> 4;
  const int ln   = lane & 15;
  const int mBase = blockIdx.y * 128 + (wave >> 2) * 64;
  const int nBase = blockIdx.x * 128 + (wave & 3) * 32;

  const f32x8 zero = {0.f, 0.f, 0.f, 0.f, 0.f, 0.f, 0.f, 0.f};
  f32x8 acc[4][2];
#pragma unroll
  for (int i = 0; i < 4; ++i)
#pragma unroll
    for (int j = 0; j < 2; ++j) acc[i][j] = zero;

  gemm_lds(X, W, blockIdx.y * 128, blockIdx.x * 128, HID / 32,
           As0, As1, Bs0, Bs1, acc);

  if (!transposeOut) {
#pragma unroll
    for (int i = 0; i < 4; ++i)
#pragma unroll
      for (int j = 0; j < 2; ++j) {
        int n = nBase + j * 16 + ln;
#pragma unroll
        for (int r = 0; r < 8; ++r) {
          int m = mBase + i * 16 + 8 * h + r;
          Y[(size_t)m * HID + n] = (bf16)acc[i][j][r];
        }
      }
  } else {
#pragma unroll
    for (int i = 0; i < 4; ++i)
#pragma unroll
      for (int j = 0; j < 2; ++j) {
        int n = nBase + j * 16 + ln;
        bf16x8 pk;
#pragma unroll
        for (int r = 0; r < 8; ++r) pk[r] = (bf16)acc[i][j][r];
        *(bf16x8*)(Yt + (size_t)n * SEQ + mBase + i * 16 + 8 * h) = pk;
      }
  }
}

// ---------------------------------------------------------------------------
// energy[SEQ,SEQ] = (Q @ K^T)/sqrt(HID), causal mask (n<=m kept). Blocks fully
// above the diagonal skip the GEMM and store NEG_INF directly.
// ---------------------------------------------------------------------------
__global__ __launch_bounds__(256) void energy_kernel(const bf16* __restrict__ Q,
                                                     const bf16* __restrict__ Kp,
                                                     float* __restrict__ att) {
  __shared__ bf16 As0[128 * PADROW], As1[128 * PADROW];
  __shared__ bf16 Bs0[128 * PADROW], Bs1[128 * PADROW];

  const int lane = threadIdx.x & 31;
  const int wave = threadIdx.x >> 5;
  const int h    = lane >> 4;
  const int ln   = lane & 15;
  const int blockM = blockIdx.y * 128;
  const int blockN = blockIdx.x * 128;
  const int mBase = blockM + (wave >> 2) * 64;
  const int nBase = blockN + (wave & 3) * 32;

  if (blockN >= blockM + 128) {  // fully masked: store NEG_INF, no GEMM/barriers
#pragma unroll
    for (int i = 0; i < 4; ++i)
#pragma unroll
      for (int j = 0; j < 2; ++j) {
        int n = nBase + j * 16 + ln;
#pragma unroll
        for (int r = 0; r < 8; ++r) {
          int m = mBase + i * 16 + 8 * h + r;
          att[(size_t)m * SEQ + n] = NEG_INF;
        }
      }
    return;
  }

  const f32x8 zero = {0.f, 0.f, 0.f, 0.f, 0.f, 0.f, 0.f, 0.f};
  f32x8 acc[4][2];
#pragma unroll
  for (int i = 0; i < 4; ++i)
#pragma unroll
    for (int j = 0; j < 2; ++j) acc[i][j] = zero;

  gemm_lds(Q, Kp, blockM, blockN, HID / 32, As0, As1, Bs0, Bs1, acc);

  const float scale = 0.03125f;  // 1/sqrt(1024)
#pragma unroll
  for (int i = 0; i < 4; ++i)
#pragma unroll
    for (int j = 0; j < 2; ++j) {
      int n = nBase + j * 16 + ln;
#pragma unroll
      for (int r = 0; r < 8; ++r) {
        int m = mBase + i * 16 + 8 * h + r;
        att[(size_t)m * SEQ + n] = (n <= m) ? acc[i][j][r] * scale : NEG_INF;
      }
    }
}

// ---------------------------------------------------------------------------
// In-place row softmax over att[SEQ,SEQ]; one workgroup per row.
// ---------------------------------------------------------------------------
__global__ __launch_bounds__(256) void softmax_kernel(float* __restrict__ att) {
  __shared__ float red[256];
  const int tid = threadIdx.x;
  float* p = att + (size_t)blockIdx.x * SEQ;

  float vals[SEQ / 256];
#pragma unroll
  for (int i = 0; i < SEQ / 256; ++i) vals[i] = p[tid + i * 256];

  float mx = vals[0];
#pragma unroll
  for (int i = 1; i < SEQ / 256; ++i) mx = fmaxf(mx, vals[i]);
  red[tid] = mx;
  __syncthreads();
  for (int s = 128; s > 0; s >>= 1) {
    if (tid < s) red[tid] = fmaxf(red[tid], red[tid + s]);
    __syncthreads();
  }
  const float rowmax = red[0];
  __syncthreads();

  float sum = 0.f;
#pragma unroll
  for (int i = 0; i < SEQ / 256; ++i) {
    vals[i] = __expf(vals[i] - rowmax);
    sum += vals[i];
  }
  red[tid] = sum;
  __syncthreads();
  for (int s = 128; s > 0; s >>= 1) {
    if (tid < s) red[tid] += red[tid + s];
    __syncthreads();
  }
  const float inv = 1.0f / red[0];
#pragma unroll
  for (int i = 0; i < SEQ / 256; ++i) p[tid + i * 256] = vals[i] * inv;
}

// ---------------------------------------------------------------------------
// x[SEQ,HID] = att[SEQ,SEQ] @ V with V supplied transposed (Vt[HID][SEQ], bf16).
// att fp32 converted to bf16 fragments on the fly; K loop stops at the diagonal
// block boundary (post-softmax masked entries are exactly 0).
// ---------------------------------------------------------------------------
__global__ __launch_bounds__(256) void av_kernel(const float* __restrict__ att,
                                                 const bf16* __restrict__ Vt,
                                                 float* __restrict__ X) {
  const int lane = threadIdx.x & 31;
  const int wave = threadIdx.x >> 5;
  const int h    = lane >> 4;
  const int ln   = lane & 15;
  const int blockM = blockIdx.y * 128;
  const int mBase = blockM + (wave >> 2) * 64;
  const int nBase = blockIdx.x * 128 + (wave & 3) * 32;
  const int kmax  = blockM + 128;

  const f32x8 zero = {0.f, 0.f, 0.f, 0.f, 0.f, 0.f, 0.f, 0.f};
  f32x8 acc[4][2];
#pragma unroll
  for (int i = 0; i < 4; ++i)
#pragma unroll
    for (int j = 0; j < 2; ++j) acc[i][j] = zero;

  for (int kbi = 0; kbi < kmax; kbi += 32) {
    const bf16* pb0 = Vt + (size_t)(nBase + ln) * SEQ + kbi + 16 * h;
    const bf16* pb1 = Vt + (size_t)(nBase + 16 + ln) * SEQ + kbi + 16 * h;
    bf16x16 b0 = frag16(pb0, pb0 + 8);
    bf16x16 b1 = frag16(pb1, pb1 + 8);
#pragma unroll
    for (int i = 0; i < 4; ++i) {
      bf16x16 a = load_a_f32(att, SEQ, mBase + i * 16 + ln, kbi, h);
      acc[i][0] = wmma_bf16(a, b0, acc[i][0]);
      acc[i][1] = wmma_bf16(a, b1, acc[i][1]);
    }
  }

#pragma unroll
  for (int i = 0; i < 4; ++i)
#pragma unroll
    for (int j = 0; j < 2; ++j) {
      int n = nBase + j * 16 + ln;
#pragma unroll
      for (int r = 0; r < 8; ++r) {
        int m = mBase + i * 16 + 8 * h + r;
        X[(size_t)m * HID + n] = acc[i][j][r];
      }
    }
}

// ---------------------------------------------------------------------------
extern "C" void kernel_launch(void* const* d_in, const int* in_sizes, int n_in,
                              void* d_out, int out_size, void* d_ws, size_t ws_size,
                              hipStream_t stream) {
  (void)in_sizes; (void)n_in; (void)out_size; (void)ws_size;

  const float* q  = (const float*)d_in[0];
  const float* k  = (const float*)d_in[1];
  const float* v  = (const float*)d_in[2];
  // d_in[3] is the tril mask; causality is applied analytically (n<=m).
  const float* Wq = (const float*)d_in[4];
  const float* Wk = (const float*)d_in[5];
  const float* Wv = (const float*)d_in[6];

  char* ws = (char*)d_ws;
  bf16* qb  = (bf16*)(ws + ((size_t)0  << 20));  // 8 MiB each
  bf16* kb  = (bf16*)(ws + ((size_t)8  << 20));
  bf16* vb  = (bf16*)(ws + ((size_t)16 << 20));
  bf16* wqb = (bf16*)(ws + ((size_t)24 << 20));  // 2 MiB each
  bf16* wkb = (bf16*)(ws + ((size_t)26 << 20));
  bf16* wvb = (bf16*)(ws + ((size_t)28 << 20));
  bf16* Qb  = (bf16*)(ws + ((size_t)30 << 20));  // 8 MiB each
  bf16* Kb  = (bf16*)(ws + ((size_t)38 << 20));
  bf16* Vt  = (bf16*)(ws + ((size_t)46 << 20));  // V^T, [HID][SEQ]

  float* x   = (float*)d_out;                       // [SEQ, HID]
  float* att = (float*)d_out + (size_t)SEQ * HID;   // [SEQ, SEQ]

  // 1) fp32 -> bf16 staging
  cvt_kernel<<<(SEQ * HID) / (256 * 4), 256, 0, stream>>>(q, qb, SEQ * HID);
  cvt_kernel<<<(SEQ * HID) / (256 * 4), 256, 0, stream>>>(k, kb, SEQ * HID);
  cvt_kernel<<<(SEQ * HID) / (256 * 4), 256, 0, stream>>>(v, vb, SEQ * HID);
  cvt_kernel<<<(HID * HID) / (256 * 4), 256, 0, stream>>>(Wq, wqb, HID * HID);
  cvt_kernel<<<(HID * HID) / (256 * 4), 256, 0, stream>>>(Wk, wkb, HID * HID);
  cvt_kernel<<<(HID * HID) / (256 * 4), 256, 0, stream>>>(Wv, wvb, HID * HID);

  // 2) projections (V written transposed)
  dim3 gp(HID / 128, SEQ / 128);
  proj_kernel<<<gp, 256, 0, stream>>>(qb, wqb, Qb, nullptr, 0);
  proj_kernel<<<gp, 256, 0, stream>>>(kb, wkb, Kb, nullptr, 0);
  proj_kernel<<<gp, 256, 0, stream>>>(vb, wvb, nullptr, Vt, 1);

  // 3) masked energy = QK^T / sqrt(HID)
  dim3 ge(SEQ / 128, SEQ / 128);
  energy_kernel<<<ge, 256, 0, stream>>>(Qb, Kb, att);

  // 4) row softmax in place (writes the `attention` output)
  softmax_kernel<<<SEQ, 256, 0, stream>>>(att);

  // 5) x = attention @ V
  dim3 ga(HID / 128, SEQ / 128);
  av_kernel<<<ga, 256, 0, stream>>>(att, Vt, x);
}
